// ICDeform_44195213476586
// MI455X (gfx1250) — matrix operand order repaired and checked by reading, not compiled, not run on hardware
//
#include <hip/hip_runtime.h>
#include <hip/hip_bf16.h>
#include <stdint.h>

typedef __attribute__((ext_vector_type(16))) __bf16 v16bf;
typedef __attribute__((ext_vector_type(8)))  __bf16 v8bf;
typedef __attribute__((ext_vector_type(8)))  float  v8f;

#define Hh   56
#define Ww   56
#define Cc   64
#define Bb   8
#define NOFF 1152        // C*2*K*K
#define KDIM 576         // C*9
#define PIX  3136        // 56*56

// ---------------- fp32 -> bf16 straight convert (for w1) ----------------
__global__ void f32_to_bf16_kernel(const float* __restrict__ in,
                                   __bf16* __restrict__ out, int n) {
  int i = blockIdx.x * 256 + threadIdx.x;
  if (i < n) out[i] = (__bf16)in[i];
}

// ---------------- im2col (NCHW fp32 -> [B][PIX][KDIM] bf16) ----------------
__global__ void im2col_bf16_kernel(const float* __restrict__ in,
                                   __bf16* __restrict__ col, int total) {
  int idx = blockIdx.x * 256 + threadIdx.x;
  if (idx >= total) return;
  int kk   = idx % KDIM;
  int rest = idx / KDIM;
  int p = rest % PIX;
  int b = rest / PIX;
  int c = kk / 9, r = kk % 9;
  int ky = r / 3, kx = r % 3;
  int h = p / Ww, w = p % Ww;
  int y = h + ky - 1, x = w + kx - 1;
  float v = 0.f;
  if (y >= 0 && y < Hh && x >= 0 && x < Ww)
    v = in[(((size_t)b * Cc + c) * Hh + y) * Ww + x];
  col[idx] = (__bf16)v;
}

// ------- conv3x3 as implicit GEMM: [PIX x KDIM] * [KDIM x NOFF] + bias -------
// Per-wave: 32x64 tile (2 A frags x 4 B frags = 8 WMMA / K-step, 8 accums).
// Block = 128 threads = 4 waves in 2(M) x 2(N); block tile 64x128.
// Grid (3136/64=49, 1152/128=9, 8).
__global__ void __launch_bounds__(128)
conv3x3_wmma_kernel(const __bf16* __restrict__ col,   // [B][PIX][KDIM]
                    const __bf16* __restrict__ wbf,   // [NOFF][KDIM]
                    const float*  __restrict__ bias,  // [NOFF]
                    float* __restrict__ out)          // [B][PIX][NOFF]
{
  const int lane  = threadIdx.x & 31;
  const int wave  = threadIdx.x >> 5;
  const int wm    = wave & 1;          // M sub-block (32 rows)
  const int wn    = wave >> 1;         // N sub-block (64 cols)
  const int b     = blockIdx.z;
  const int Mbase = blockIdx.x * 64 + wm * 32;
  const int Nbase = blockIdx.y * 128 + wn * 64;

  // A fragment addressing: lane holds row M = Mtile + (lane&15),
  // K elements kb..kb+15, kb = 16*(lane>=16).
  const int kb = (lane >> 4) * 16;
  const __bf16* aptr0 =
      col + ((size_t)b * PIX + Mbase + (lane & 15)) * KDIM + kb;
  const __bf16* aptr1 = aptr0 + (size_t)16 * KDIM;

  // B fragment addressing: lane holds column N = Nbase + j*16 + (lane&15),
  // K elements krb..krb+15, krb = 16*(lane>=16).
  const int ncol = lane & 15;
  const int krb  = (lane >> 4) * 16;
  const __bf16* bptr[4];
  #pragma unroll
  for (int j = 0; j < 4; ++j)
    bptr[j] = wbf + (size_t)(Nbase + j * 16 + ncol) * KDIM + krb;

  v8f acc[2][4];
  #pragma unroll
  for (int i = 0; i < 2; ++i)
    #pragma unroll
    for (int j = 0; j < 4; ++j) acc[i][j] = (v8f){};

  for (int ks = 0; ks < KDIM / 32; ++ks) {
    const int ko = ks * 32;

    v16bf afrag[2];
    {
      v8bf a0lo = *(const v8bf*)(aptr0 + ko);
      v8bf a0hi = *(const v8bf*)(aptr0 + ko + 8);
      v8bf a1lo = *(const v8bf*)(aptr1 + ko);
      v8bf a1hi = *(const v8bf*)(aptr1 + ko + 8);
      #pragma unroll
      for (int i = 0; i < 8; ++i) {
        afrag[0][i] = a0lo[i]; afrag[0][i + 8] = a0hi[i];
        afrag[1][i] = a1lo[i]; afrag[1][i + 8] = a1hi[i];
      }
    }

    #pragma unroll
    for (int j = 0; j < 4; ++j) {
      v16bf bf = *(const v16bf*)(bptr[j] + ko);
      acc[0][j] = __builtin_amdgcn_wmma_f32_16x16x32_bf16(
          false, afrag[0], false, bf, (short)0, acc[0][j], false, false);
      acc[1][j] = __builtin_amdgcn_wmma_f32_16x16x32_bf16(
          false, afrag[1], false, bf, (short)0, acc[1][j], false, false);
    }
  }

  // C/D layout: VGPR v, lane l -> M = v + 8*(l>=16), N = l&15.
  const int prow = Mbase + 8 * (lane >> 4);
  float* outp = out + ((size_t)b * PIX + prow) * NOFF + Nbase + ncol;
  float bi[4];
  #pragma unroll
  for (int j = 0; j < 4; ++j) bi[j] = bias[Nbase + j * 16 + ncol];

  #pragma unroll
  for (int i = 0; i < 2; ++i) {
    float* rowbase = outp + (size_t)i * 16 * NOFF;
    #pragma unroll
    for (int v = 0; v < 8; ++v) {
      float* r = rowbase + (size_t)v * NOFF;
      #pragma unroll
      for (int j = 0; j < 4; ++j) r[j * 16] = acc[i][j][v] + bi[j];
    }
  }
}

// ---------------- deformable conv: bilinear gather + 576-way contraction ---
// One thread per output pixel (b,h,w); 64 fp32 accumulators in VGPRs.
__global__ void __launch_bounds__(256)
deform_kernel(const float* __restrict__ x,    // [B][C][H][W]
              const float* __restrict__ off,  // [B][PIX][NOFF] (NHWC)
              const float* __restrict__ wd,   // [C][C][9]
              float* __restrict__ out)        // [B][C][PIX]
{
  int tid = blockIdx.x * 256 + threadIdx.x;
  if (tid >= Bb * PIX) return;
  int b = tid / PIX, p = tid % PIX;
  int h = p / Ww, w = p % Ww;

  float acc[Cc];
  #pragma unroll
  for (int o = 0; o < Cc; ++o) acc[o] = 0.f;

  const float* offp = off + ((size_t)b * PIX + p) * NOFF;
  for (int c = 0; c < Cc; ++c) {
    const float* xc = x + ((size_t)b * Cc + c) * PIX;
    #pragma unroll 3
    for (int k = 0; k < 9; ++k) {
      int ky = k / 3, kx = k % 3;
      float dy = offp[c * 18 + k * 2 + 0];
      float dx = offp[c * 18 + k * 2 + 1];
      float py = (float)(h - 1 + ky) + dy;
      float px = (float)(w - 1 + kx) + dx;
      float y0f = floorf(py), x0f = floorf(px);
      float wy1 = py - y0f, wx1 = px - x0f;
      float wy0 = 1.f - wy1, wx0 = 1.f - wx1;
      int y0 = (int)y0f, x0 = (int)x0f;

      float t00 = 0.f, t01 = 0.f, t10 = 0.f, t11 = 0.f;
      if (y0 >= 0 && y0 < Hh) {
        if (x0 >= 0     && x0 < Ww)     t00 = xc[y0 * Ww + x0];
        if (x0 + 1 >= 0 && x0 + 1 < Ww) t01 = xc[y0 * Ww + x0 + 1];
      }
      if (y0 + 1 >= 0 && y0 + 1 < Hh) {
        if (x0 >= 0     && x0 < Ww)     t10 = xc[(y0 + 1) * Ww + x0];
        if (x0 + 1 >= 0 && x0 + 1 < Ww) t11 = xc[(y0 + 1) * Ww + x0 + 1];
      }
      float s = wy0 * wx0 * t00 + wy0 * wx1 * t01
              + wy1 * wx0 * t10 + wy1 * wx1 * t11;

      const float* wp = wd + c * 9 + k;  // uniform across lanes -> s_load
      #pragma unroll
      for (int o = 0; o < Cc; ++o)
        acc[o] = fmaf(s, wp[(size_t)o * KDIM], acc[o]);
    }
  }
  float* op = out + (size_t)b * Cc * PIX + p;
  #pragma unroll
  for (int o = 0; o < Cc; ++o) op[(size_t)o * PIX] = acc[o];
}

// ---------------------------------------------------------------------------
extern "C" void kernel_launch(void* const* d_in, const int* in_sizes, int n_in,
                              void* d_out, int out_size, void* d_ws, size_t ws_size,
                              hipStream_t stream) {
  const float* x  = (const float*)d_in[0];   // (8,64,56,56)
  const float* w1 = (const float*)d_in[1];   // (1152,64,3,3)
  const float* b1 = (const float*)d_in[2];   // (1152,)
  const float* wd = (const float*)d_in[3];   // (64,64,3,3)
  float* out = (float*)d_out;                // (8,64,56,56)

  char* ws = (char*)d_ws;
  const size_t W1BF_BYTES = (size_t)NOFF * KDIM * 2;          //  1.33 MB
  const size_t COL_BYTES  = (size_t)Bb * PIX * KDIM * 2;      // 28.9  MB
  const size_t OFF_BYTES  = (size_t)Bb * PIX * NOFF * 4;      // 115.6 MB
  __bf16* w1bf = (__bf16*)ws;
  __bf16* col  = (__bf16*)(ws + W1BF_BYTES);
  float*  offb = (float*) (ws + W1BF_BYTES + COL_BYTES);
  float*  tbuf = (float*) (ws + W1BF_BYTES + COL_BYTES + OFF_BYTES);

  // 1. weights -> bf16 (layout already [n][c*9+r] row-major = GEMM B)
  { int n = NOFF * KDIM;
    f32_to_bf16_kernel<<<(n + 255) / 256, 256, 0, stream>>>(w1, w1bf, n); }

  // 2. im2col(x) -> bf16
  { int n = Bb * PIX * KDIM;
    im2col_bf16_kernel<<<(n + 255) / 256, 256, 0, stream>>>(x, col, n); }

  // 3. off1 = conv3x3(x) via WMMA
  { dim3 g(PIX / 64, NOFF / 128, Bb);
    conv3x3_wmma_kernel<<<g, 128, 0, stream>>>(col, w1bf, b1, offb); }

  // 4. t = deform_conv(x, off1)
  { int n = Bb * PIX;
    deform_kernel<<<(n + 255) / 256, 256, 0, stream>>>(x, offb, wd, tbuf); }

  // 5. im2col(t)
  { int n = Bb * PIX * KDIM;
    im2col_bf16_kernel<<<(n + 255) / 256, 256, 0, stream>>>(tbuf, col, n); }

  // 6. off2 = conv3x3(t) via WMMA
  { dim3 g(PIX / 64, NOFF / 128, Bb);
    conv3x3_wmma_kernel<<<g, 128, 0, stream>>>(col, w1bf, b1, offb); }

  // 7. out = deform_conv(x, off2)
  { int n = Bb * PIX;
    deform_kernel<<<(n + 255) / 256, 256, 0, stream>>>(x, offb, wd, out); }
}
